// Correlation_19799799234876
// MI455X (gfx1250) — matrix operand order repaired and checked by reading, not compiled, not run on hardware
//
#include <hip/hip_runtime.h>

typedef _Float16 h8  __attribute__((ext_vector_type(8)));
typedef _Float16 h16 __attribute__((ext_vector_type(16)));
typedef float    f8  __attribute__((ext_vector_type(8)));
typedef float    f4  __attribute__((ext_vector_type(4)));
typedef float    f2  __attribute__((ext_vector_type(2)));

#define BATCH 8
#define CC    128
#define HH    128
#define WW    256
#define HW    (HH * WW)
#define ND    9
#define SSTR  136                   // halves per x-row (272B, 16B aligned)
#define S1ROW (256 * SSTR)          // halves per staged in1 image-row
#define SOSTR 260                   // floats per d-row (1040B, 16B aligned)
#define SOROW 10                    // 9 dx rows + 1 scratch
#define SOHALF (SOROW * SOSTR)
#define S2BYTES (272 * SSTR * 2)    // 73984 B
#define SMEM_BYTES (2 * S1ROW * 2)  // 139264 B (two in1 rows; overlays s2+so)

__global__ __launch_bounds__(1024, 1)
void corr_wmma_kernel(const float* __restrict__ in1,
                      const float* __restrict__ in2,
                      float* __restrict__ out) {
  __shared__ __align__(16) unsigned char smem[SMEM_BYTES];
  _Float16* s1 = (_Float16*)smem;               // phase 1: s1[row][x][c] f16
  _Float16* s2 = (_Float16*)smem;               // phase 2: s2[x+4][c] f16
  float*    so = (float*)(smem + S2BYTES);      // phase 2: out stage, 2 halves

  const int tid  = threadIdx.x;
  const int b    = blockIdx.x >> 6;
  const int y    = (blockIdx.x & 63) * 2;       // even; y+1 always < HH
  const int lane = tid & 31;
  const int wv   = tid >> 5;                    // 0..31
  const int half = wv >> 4;                     // 0: row y, 1: row y+1
  const int x0   = (wv & 15) * 16;
  const int thid = tid & 511;                   // id within half-block
  const int yo   = y + half;

  const float scale = 1.0f / (float)CC;         // folded into in1 (exact: 2^-7)

  // ------- stage 2 in1 rows: float2 along x, 8 channels packed, pre-scaled --
  {
    const float* src = in1 + ((size_t)(b * CC) * HH + y) * WW;
    #pragma unroll
    for (int it = 0; it < 4; ++it) {            // 2 rows * 128 xp * 16 cg = 4096
      int e   = tid + it * 1024;
      int r   = e >> 11;
      int rem = e & 2047;
      int x   = (rem & 127) * 2;
      int c0  = (rem >> 7) * 8;
      const float* s = src + (size_t)r * WW + x;
      h8 va, vb;
      #pragma unroll
      for (int j = 0; j < 8; ++j) {
        f2 v = *(const f2*)(s + (size_t)(c0 + j) * HW);
        va[j] = (_Float16)(v.x * scale);
        vb[j] = (_Float16)(v.y * scale);
      }
      _Float16* dst = s1 + ((r << 8) + x) * SSTR + c0;
      *(h8*)dst          = va;
      *(h8*)(dst + SSTR) = vb;
    }
  }
  __syncthreads();

  // ------------- gather A fragments from own row (resident in VGPRs) -------
  const int hi  = lane >> 4;
  const int l15 = lane & 15;
  h16 afrag[4];
  {
    const _Float16* arow = s1 + (size_t)half * S1ROW + (x0 + l15) * SSTR;
    #pragma unroll
    for (int k = 0; k < 4; ++k) {
      // A 16x32 f16 layout: e=0..7 -> K=8hi+e ; e=8..15 -> K=16+8hi+(e-8)
      const _Float16* p = arow + 32 * k + 8 * hi;
      h8 lo = *(const h8*)(p);
      h8 hh = *(const h8*)(p + 16);
      afrag[k] = __builtin_shufflevector(lo, hh,
                   0,1,2,3,4,5,6,7,8,9,10,11,12,13,14,15);
    }
  }
  // s1 is dead now; its space is reused as s2+so. The first in-range
  // iteration's leading __syncthreads orders these reads before overwrites.

  for (int i = 0; i < 10; ++i) {                // staged in2 rows y-4 .. y+5
    const int  y2      = y - 4 + i;
    const bool inRange = (y2 >= 0) && (y2 < HH);          // block-uniform
    const bool myValid = ((unsigned)(i - half) <= 8u);    // wave-uniform

    if (inRange) {
      __syncthreads();   // prev: s2 gathers + so flush reads complete
      const float* src2 = in2 + ((size_t)(b * CC) * HH + y2) * WW;

      // prefetch next staged row (128c x 9 lines of 32 floats)
      if (i < 9 && y2 + 1 < HH) {
        const float* src3 = src2 + WW;
        #pragma unroll
        for (int it = 0; it < 2; ++it) {
          int e = tid + it * 1024;
          if (e < 1152) {                       // wave-uniform guard
            int c  = e / 9;
            int xl = (e - c * 9) * 32 - 4;
            __builtin_prefetch(src3 + (size_t)c * HW + xl, 0, 3);
          }
        }
      }

      // stage in2 row y2 (halo x in [-4,268)): float2 pairs, 8 ch packed
      #pragma unroll
      for (int it = 0; it < 3; ++it) {          // 136 xp * 16 cg = 2176
        int e = tid + it * 1024;
        if (e < 2176) {                         // wave-uniform guard
          int cg = e / 136;
          int x  = (e - cg * 136) * 2;          // 0..270 even
          int c0 = cg * 8;
          int xg = x - 4;                       // pair fully valid or fully OOB
          h8 va, vb;
          if ((unsigned)xg < 256u) {
            const float* s = src2 + xg;
            #pragma unroll
            for (int j = 0; j < 8; ++j) {
              f2 v = *(const f2*)(s + (size_t)(c0 + j) * HW);
              va[j] = (_Float16)v.x;
              vb[j] = (_Float16)v.y;
            }
          } else {
            #pragma unroll
            for (int j = 0; j < 8; ++j) { va[j] = (_Float16)0.f; vb[j] = (_Float16)0.f; }
          }
          _Float16* dst = s2 + x * SSTR + c0;
          *(h8*)dst          = va;
          *(h8*)(dst + SSTR) = vb;
        }
      }
      __syncthreads();

      if (myValid) {
        float* soH = so + half * SOHALF;
        // Interleaved dual-tile WMMA streams: independent accumulators so
        // one s_wait_dscnt covers two WMMAs and next-k loads overlap compute.
        const _Float16* brow0 = s2 + (x0 + l15) * SSTR + 8 * hi;
        const _Float16* brow1 = brow0 + 16 * SSTR;
        f8 acc0 = {0.f, 0.f, 0.f, 0.f, 0.f, 0.f, 0.f, 0.f};
        f8 acc1 = {0.f, 0.f, 0.f, 0.f, 0.f, 0.f, 0.f, 0.f};
        #pragma unroll
        for (int k = 0; k < 4; ++k) {
          // B 32x16 f16 layout: e=0..7 -> K=8hi+e ; e=8..15 -> K=16+8hi+(e-8)
          const _Float16* p0 = brow0 + 32 * k;
          const _Float16* p1 = brow1 + 32 * k;
          h8 l0 = *(const h8*)(p0);
          h8 h0 = *(const h8*)(p0 + 16);
          h8 l1 = *(const h8*)(p1);
          h8 h1 = *(const h8*)(p1 + 16);
          h16 b0 = __builtin_shufflevector(l0, h0,
                     0,1,2,3,4,5,6,7,8,9,10,11,12,13,14,15);
          h16 b1 = __builtin_shufflevector(l1, h1,
                     0,1,2,3,4,5,6,7,8,9,10,11,12,13,14,15);
          acc0 = __builtin_amdgcn_wmma_f32_16x16x32_f16(
                   false, afrag[k], false, b0, (short)0, acc0, false, false);
          acc1 = __builtin_amdgcn_wmma_f32_16x16x32_f16(
                   false, afrag[k], false, b1, (short)0, acc1, false, false);
        }
        // branch-free scatter: D[m][n] -> so[dx+4][x0+m], invalid -> row 9
        #pragma unroll
        for (int r = 0; r < 8; ++r) {
          int m     = r + 8 * hi;
          int dsel0 = l15 - m;                  // dx+4, tile 0
          int dsel1 = dsel0 + 16;               // dx+4, tile 1
          int row0  = ((unsigned)dsel0 <= 8u) ? dsel0 : 9;
          int row1  = ((unsigned)dsel1 <= 8u) ? dsel1 : 9;
          soH[row0 * SOSTR + x0 + m] = acc0[r];
          soH[row1 * SOSTR + x0 + m] = acc1[r];
        }
      }
      __syncthreads();   // so visible to the flush below
    }

    // flush my half's 9x256 slice (zeros if the staged row is padding)
    if (myValid) {
      const int dbase = (i - half) * ND;        // (dy+4)*9
      float* outd = out + ((size_t)(b * 81 + dbase) * HH + yo) * WW;
      if (inRange) {
        const float* soH = so + half * SOHALF;
        #pragma unroll
        for (int it = 0; it < 2; ++it) {
          int e = thid + it * 512;
          if (e < 576) {                        // 9*64 float4s, wave-uniform
            int d  = e >> 6;
            int xq = (e & 63) * 4;
            f4 v = *(const f4*)&soH[d * SOSTR + xq];
            *(f4*)&outd[(size_t)d * HW + xq] = v;
          }
        }
      } else {
        #pragma unroll
        for (int it = 0; it < 2; ++it) {
          int e = thid + it * 512;
          if (e < 576) {
            int d  = e >> 6;
            int xq = (e & 63) * 4;
            f4 z = {0.f, 0.f, 0.f, 0.f};
            *(f4*)&outd[(size_t)d * HW + xq] = z;
          }
        }
      }
    }
  }
}

extern "C" void kernel_launch(void* const* d_in, const int* in_sizes, int n_in,
                              void* d_out, int out_size, void* d_ws, size_t ws_size,
                              hipStream_t stream) {
  const float* in1 = (const float*)d_in[0];
  const float* in2 = (const float*)d_in[1];
  float* out = (float*)d_out;
  (void)in_sizes; (void)n_in; (void)out_size; (void)d_ws; (void)ws_size;

  dim3 grid(BATCH * (HH / 2));   // one workgroup per (b, row-pair)
  dim3 block(1024);              // 32 wave32 waves; 16 per output row
  corr_wmma_kernel<<<grid, block, 0, stream>>>(in1, in2, out);
}
